// BDESN_80281528697358
// MI455X (gfx1250) — matrix operand order repaired and checked by reading, not compiled, not run on hardware
//
#include <hip/hip_runtime.h>
#include <hip/hip_bf16.h>

// ---------------------------------------------------------------------------
// BDESN on MI455X (gfx1250):
//   prep_kernel    : w (f32->bf16), w_out^T (f32->bf16), barrier reset
//   uproj_kernel   : u_proj[t][b][h] = u[b,t,:].w_in[h,:] + bias[h]   (f32)
//   scan_kernel    : persistent, 16 WGs, fused fwd+bwd leaky-tanh scan,
//                    v_wmma_f32_16x16x32_bf16, weights pinned in LDS,
//                    h staged via GLOBAL_LOAD_ASYNC_TO_LDS_B128 (ASYNCcnt),
//                    one global barrier per time step
//   readout_kernel : out = bias + states @ w_out[1:]   (WMMA GEMM, N=16)
// ---------------------------------------------------------------------------

#define BB 8
#define TT 2048
#define DD 64
#define HH 1024
#define OO 16
#define NWG 16           // workgroups in the scan (each owns 64 cols of H)
#define SCAN_THREADS 128 // 4 waves -> 4 N-tiles of 16 per WG
static constexpr float ALPHA = 0.9f;

typedef __attribute__((ext_vector_type(16))) __bf16 v16bf;
typedef __attribute__((ext_vector_type(8)))  float  v8f;
typedef __attribute__((ext_vector_type(4)))  unsigned int u32x4;

union Frag16 {           // one WMMA 16-bit A/B operand: 32 bytes = 2 x b128
    v16bf v;
    u32x4 q[2];
};

__device__ __forceinline__ unsigned short f2bf(float f) {
    unsigned u = __float_as_uint(f);
    u += 0x7FFFu + ((u >> 16) & 1u);      // round-to-nearest-even
    return (unsigned short)(u >> 16);
}
__device__ __forceinline__ float bf2f(unsigned short b) {
    return __uint_as_float(((unsigned)b) << 16);
}

// Async global->LDS b128 (CDNA5 TDM-adjacent path, tracked by ASYNCcnt).
// Per the ISA aperture rule, the low 32 bits
// of a generic shared-memory pointer are the per-lane LDS destination.
__device__ __forceinline__ void async_copy_b128(void* lds_dst, const void* gsrc) {
    unsigned ldst = (unsigned)(size_t)lds_dst;
    asm volatile("global_load_async_to_lds_b128 %0, %1, off"
                 :: "v"(ldst), "v"(gsrc) : "memory");
}
__device__ __forceinline__ void wait_asynccnt0() {
    asm volatile("s_wait_asynccnt 0x0" ::: "memory");
}

// ---------------------------------------------------------------------------
// prep: w -> bf16 [H][H]; w_out rows 1..2H transposed -> bf16 [O][2H]; cnt=0
// ---------------------------------------------------------------------------
__global__ void prep_kernel(const float* __restrict__ w,
                            const float* __restrict__ w_out,
                            unsigned short* __restrict__ w_bf,
                            unsigned short* __restrict__ woT,
                            unsigned* __restrict__ barrier_cnt) {
    int i = blockIdx.x * 256 + threadIdx.x;
    int stride = gridDim.x * 256;
    for (int idx = i; idx < HH * HH; idx += stride)
        w_bf[idx] = f2bf(w[idx]);
    for (int idx = i; idx < OO * 2 * HH; idx += stride) {
        int o = idx / (2 * HH), k = idx % (2 * HH);
        woT[idx] = f2bf(w_out[(size_t)(1 + k) * OO + o]); // woT[o][k]
    }
    if (i == 0) *barrier_cnt = 0u;
}

// ---------------------------------------------------------------------------
// input projection: grid (T, H/256), block 256.  K=64 in registers.
// ---------------------------------------------------------------------------
__global__ void uproj_kernel(const float* __restrict__ u,
                             const float* __restrict__ w_in,
                             const float* __restrict__ w_bias,
                             float* __restrict__ uproj) {
    __shared__ float ulds[BB][DD];
    const int t = blockIdx.x;
    const int tid = threadIdx.x;
    for (int i = tid; i < BB * DD; i += 256) {
        int b = i / DD, d = i % DD;
        ulds[b][d] = u[((size_t)b * TT + t) * DD + d];
    }
    __syncthreads();
    const int h = blockIdx.y * 256 + tid;
    const float* wr = w_in + (size_t)h * DD;
    float wreg[DD];
#pragma unroll
    for (int d = 0; d < DD; ++d) wreg[d] = wr[d];
    const float bias = w_bias[h];
    for (int b = 0; b < BB; ++b) {
        float acc = bias;
#pragma unroll
        for (int d = 0; d < DD; ++d) acc += ulds[b][d] * wreg[d];
        uproj[((size_t)t * BB + b) * HH + h] = acc;
    }
}

// ---------------------------------------------------------------------------
// the sequential scan.  16 persistent WGs x 128 threads (4 waves).
// LDS: w slice [64][1024] bf16 (128 KB, loaded once) + h_cur [16][1024] bf16.
// A-matrix rows 0..7 = h_fwd(batch), 8..15 = h_bwd(batch).
// ---------------------------------------------------------------------------
__global__ void scan_kernel(const float* __restrict__ uproj,
                            const unsigned short* __restrict__ w_bf,
                            unsigned short* __restrict__ states,   // [B][T][2H] bf16
                            unsigned* __restrict__ barrier_cnt) {
    extern __shared__ char smem[];
    unsigned short* w_lds = (unsigned short*)smem;                    // [64][HH]
    unsigned short* h_lds = (unsigned short*)(smem + 64 * HH * 2);    // [16][HH]

    const int wg   = blockIdx.x;           // 0..15
    const int tid  = threadIdx.x;          // 0..127
    const int lane = tid & 31;
    const int wave = tid >> 5;             // 0..3
    const int n0g  = wg * 64;              // global col base of this WG

    // --- load this WG's 64 rows of w into LDS (once; reused for 2048 steps)
    {
        const u32x4* src = (const u32x4*)(w_bf + (size_t)n0g * HH);
        u32x4* dst = (u32x4*)w_lds;
        for (int i = tid; i < 64 * HH / 8; i += SCAN_THREADS) dst[i] = src[i];
    }
    __syncthreads();

    const int lm    = lane & 15;           // tile row (A) / tile col (B,C)
    const int khalf = lane >> 4;
    const int nl    = wave * 16;           // local col base within slice
    const int ng    = n0g + nl + lm;       // this lane's global output column

    for (int s = 0; s < TT; ++s) {
        // ---- stage h(prev) [16][1024] into LDS ------------------------------
        if (s == 0) {
            u32x4 z = {0u, 0u, 0u, 0u};
            u32x4* dst = (u32x4*)h_lds;
            for (int i = tid; i < 16 * HH / 8; i += SCAN_THREADS) dst[i] = z;
        } else {
            // direct global->LDS async copies: no VGPR round trip, one
            // s_wait_asynccnt 0 covers all 16 chunks issued by this thread
            for (int i = tid; i < 16 * HH / 8; i += SCAN_THREADS) {
                int row = i / (HH / 8), cv = i % (HH / 8);
                const unsigned short* srcrow;
                if (row < 8)   // fwd batch `row`, time s-1, cols 0..H-1
                    srcrow = states + ((size_t)row * TT + (s - 1)) * (2 * HH);
                else           // bwd batch row-8, time T-s, cols H..2H-1
                    srcrow = states + ((size_t)(row - 8) * TT + (TT - s)) * (2 * HH) + HH;
                async_copy_b128(h_lds + (size_t)i * 8, srcrow + (size_t)cv * 8);
            }
            wait_asynccnt0();
        }
        __syncthreads();

        // ---- C[16x16] = h_cur[16xH] . w_slice^T, two interleaved accumulators
        v8f c0 = {};
        v8f c1 = {};
        if (s > 0) {
            const unsigned short* hrow = h_lds + (size_t)lm * HH;
            const unsigned short* wrow = w_lds + (size_t)(nl + lm) * HH;
            for (int k0 = 0; k0 < HH; k0 += 64) {
                Frag16 a0, b0, a1, b1;
                const u32x4* ha = (const u32x4*)(hrow + k0);
                a0.q[0] = ha[2 * khalf];     a0.q[1] = ha[2 * khalf + 1];
                a1.q[0] = ha[4 + 2 * khalf]; a1.q[1] = ha[5 + 2 * khalf];
                const u32x4* wa = (const u32x4*)(wrow + k0 + khalf * 16);
                b0.q[0] = wa[0]; b0.q[1] = wa[1];
                b1.q[0] = wa[4]; b1.q[1] = wa[5];
                c0 = __builtin_amdgcn_wmma_f32_16x16x32_bf16(
                        false, a0.v, false, b0.v, (short)0, c0, false, false);
                c1 = __builtin_amdgcn_wmma_f32_16x16x32_bf16(
                        false, a1.v, false, b1.v, (short)0, c1, false, false);
            }
        }

        // ---- leaky tanh update + scatter into states ------------------------
#pragma unroll
        for (int r = 0; r < 8; ++r) {
            const int m = r + khalf * 8;              // C row (ISA layout)
            const float acc = c0[r] + c1[r];
            float up;
            size_t sidx;
            if (m < 8) {                              // forward, batch m, time s
                up   = uproj[((size_t)s * BB + m) * HH + ng];
                sidx = ((size_t)m * TT + s) * (2 * HH) + ng;
            } else {                                  // backward, batch m-8, time T-1-s
                const int b  = m - 8;
                const int tb = TT - 1 - s;
                up   = uproj[((size_t)tb * BB + b) * HH + ng];
                sidx = ((size_t)b * TT + tb) * (2 * HH) + HH + ng;
            }
            const float prev = bf2f(h_lds[(size_t)m * HH + ng]); // zero at s==0
            const float hn   = (1.0f - ALPHA) * prev + ALPHA * tanhf(up + acc);
            states[sidx] = f2bf(hn);
        }

        // ---- device-wide barrier (monotonic count, 16 co-resident WGs) ------
        __threadfence();
        __syncthreads();
        if (tid == 0) {
            atomicAdd(barrier_cnt, 1u);
            const unsigned target = (unsigned)(s + 1) * NWG;
            while (__hip_atomic_load(barrier_cnt, __ATOMIC_ACQUIRE,
                                     __HIP_MEMORY_SCOPE_AGENT) < target) {
                __builtin_amdgcn_s_sleep(2);
            }
        }
        __syncthreads();
    }
}

// ---------------------------------------------------------------------------
// readout: out[row][o] = w_out[0][o] + states[row][:] . w_out[1:][:,o]
// 1024 M-tiles, 8 waves/block (1 tile each), K=2048 -> 64 WMMAs per wave.
// ---------------------------------------------------------------------------
__global__ void readout_kernel(const unsigned short* __restrict__ states,
                               const unsigned short* __restrict__ woT, // [O][2H] bf16
                               const float* __restrict__ w_out,
                               float* __restrict__ out) {
    extern __shared__ char smem[];
    unsigned short* wo_lds = (unsigned short*)smem;  // [16][2H] = 64 KB
    const int tid = threadIdx.x, lane = tid & 31, wave = tid >> 5;
    for (int i = tid; i < OO * 2 * HH / 8; i += 256)
        ((u32x4*)wo_lds)[i] = ((const u32x4*)woT)[i];
    __syncthreads();

    const int mtile = blockIdx.x * 8 + wave;         // 0..1023
    const int lm = lane & 15, khalf = lane >> 4;
    const unsigned short* arow = states + ((size_t)(mtile * 16 + lm)) * (2 * HH);
    const unsigned short* brow = wo_lds + (size_t)lm * (2 * HH);

    v8f c0 = {};
    v8f c1 = {};
    for (int k0 = 0; k0 < 2 * HH; k0 += 64) {
        Frag16 a0, b0, a1, b1;
        const u32x4* ha = (const u32x4*)(arow + k0);
        a0.q[0] = ha[2 * khalf];     a0.q[1] = ha[2 * khalf + 1];
        a1.q[0] = ha[4 + 2 * khalf]; a1.q[1] = ha[5 + 2 * khalf];
        const u32x4* wa = (const u32x4*)(brow + k0 + khalf * 16);
        b0.q[0] = wa[0]; b0.q[1] = wa[1];
        b1.q[0] = wa[4]; b1.q[1] = wa[5];
        c0 = __builtin_amdgcn_wmma_f32_16x16x32_bf16(
                false, a0.v, false, b0.v, (short)0, c0, false, false);
        c1 = __builtin_amdgcn_wmma_f32_16x16x32_bf16(
                false, a1.v, false, b1.v, (short)0, c1, false, false);
    }
    const float biaso = w_out[lm];                   // w_out[0][o], o = lm
#pragma unroll
    for (int r = 0; r < 8; ++r) {
        const int row = mtile * 16 + r + khalf * 8;
        out[(size_t)row * OO + lm] = c0[r] + c1[r] + biaso;
    }
}

// ---------------------------------------------------------------------------
extern "C" void kernel_launch(void* const* d_in, const int* in_sizes, int n_in,
                              void* d_out, int out_size, void* d_ws, size_t ws_size,
                              hipStream_t stream) {
    const float* u      = (const float*)d_in[0];   // [B,T,D]
    const float* w      = (const float*)d_in[1];   // [H,H]
    const float* w_in   = (const float*)d_in[2];   // [H,D]
    const float* w_bias = (const float*)d_in[3];   // [H]
    const float* w_out  = (const float*)d_in[4];   // [2H+1, O]
    float* out = (float*)d_out;                    // [B,T,O]

    // workspace layout
    char* ws = (char*)d_ws;
    unsigned short* w_bf  = (unsigned short*)ws;                         // 2 MB
    unsigned short* woT   = (unsigned short*)(ws + (size_t)HH * HH * 2); // 64 KB
    unsigned* barrier_cnt = (unsigned*)(ws + (size_t)HH * HH * 2 + (size_t)OO * 2 * HH * 2);
    float* uproj = (float*)(ws + 2163200);                               // 64 MB, 256-aligned
    unsigned short* st = (unsigned short*)(ws + 2163200 + (size_t)TT * BB * HH * 4); // 64 MB

    prep_kernel<<<256, 256, 0, stream>>>(w, w_out, w_bf, woT, barrier_cnt);
    uproj_kernel<<<dim3(TT, HH / 256), 256, 0, stream>>>(u, w_in, w_bias, uproj);
    scan_kernel<<<NWG, SCAN_THREADS, (64 * HH + 16 * HH) * 2, stream>>>(
        uproj, w_bf, st, barrier_cnt);
    readout_kernel<<<(BB * TT / 16) / 8, 256, OO * 2 * HH * 2, stream>>>(
        st, woT, w_out, out);
}
